// WindowGRU_32976758898906
// MI455X (gfx1250) — matrix operand order repaired and checked by reading, not compiled, not run on hardware
//
#include <hip/hip_runtime.h>
#include <hip/hip_bf16.h>

typedef __bf16 v16bf __attribute__((ext_vector_type(16)));
typedef float  v8f   __attribute__((ext_vector_type(8)));
typedef float  f32x4 __attribute__((ext_vector_type(4)));
typedef int    v4i   __attribute__((ext_vector_type(4)));

#define B_  128
#define S_  1024
#define I_  768
#define H_  256
#define O_  6
#define G3  768   // 3*H

union FragU {
    v16bf  v;      // one WMMA 16x32 bf16 operand (per-lane 16 elems = 8 VGPRs)
    f32x4  f4[2];  // two 16-byte chunks for b128 load/store
    __bf16 h[16];
};

// address-space-qualified int4 pointers for the async-to-LDS builtin
#define AS1I4(p) ((__attribute__((address_space(1))) v4i*)(unsigned long long)(p))
#define AS3I4(p) ((__attribute__((address_space(3))) v4i*)(unsigned int)(unsigned long long)(p))

__device__ __forceinline__ void async_b128(const void* g, void* l) {
    // GLOBAL_LOAD_ASYNC_TO_LDS_B128: memory -> LDS, tracked by ASYNCcnt
    __builtin_amdgcn_global_load_async_to_lds_b128(AS1I4(g), AS3I4(l), 0, 0);
}
__device__ __forceinline__ void wait_async0() {
    asm volatile("s_wait_asynccnt 0x0" ::: "memory");
}

__device__ __forceinline__ float sigmoid_f(float x) {
    return __builtin_amdgcn_rcpf(1.0f + __expf(-x));
}
__device__ __forceinline__ float tanh_f(float x) {
    // 1 - 2/(e^{2x}+1): saturates correctly at +/-1
    return 1.0f - 2.0f * __builtin_amdgcn_rcpf(__expf(2.0f * x) + 1.0f);
}

// ---------------------------------------------------------------------------
// Prep B: convert row-major f32 weights [G rows x K cols] into bf16 WMMA
// B-fragments, fragment-linear: frag f = nt*nFragK + kc is a 32(K)x16(N)
// tile; lane L holds 16 contiguous bf16 (32B) at f*512 + L*16.
// Lane L<16: column N=L, K={k0..k0+7, k0+16..k0+23} ; lane>=16: same col, K+8.
// ---------------------------------------------------------------------------
__global__ void prep_w_bf16(const float* __restrict__ W, __bf16* __restrict__ out,
                            int K, int nFragK, int nFrag) {
    int gid = blockIdx.x * blockDim.x + threadIdx.x;
    int f = gid >> 5;
    if (f >= nFrag) return;
    int l  = gid & 31;
    int nt = f / nFragK;
    int kc = f - nt * nFragK;
    int g  = nt * 16 + (l & 15);
    int k0 = kc * 32 + (l >> 4) * 8;
    const f32x4* src = reinterpret_cast<const f32x4*>(W + (size_t)g * K + k0);
    float buf[16];
#pragma unroll
    for (int q = 0; q < 4; ++q) {
        f32x4 v = src[(q & 1) + (q >> 1) * 4];
        buf[q*4+0] = v.x; buf[q*4+1] = v.y; buf[q*4+2] = v.z; buf[q*4+3] = v.w;
    }
    FragU fr;
#pragma unroll
    for (int j = 0; j < 16; ++j) fr.h[j] = (__bf16)buf[j];
    f32x4* dst = reinterpret_cast<f32x4*>(out + (size_t)f * 512 + l * 16);
    dst[0] = fr.f4[0];
    dst[1] = fr.f4[1];
}

// ---------------------------------------------------------------------------
// Prep A: convert x [B*S rows x I cols] f32 into bf16 WMMA A-fragments,
// fragment-linear: frag f = mtile*24 + kc (mtile = 16-row group, kc = 32-k
// chunk); lane L holds its 16 A elements as 32 contiguous bytes.
// A 16x32 layout: lane<16 -> row=L, K={k0..k0+7, k0+16..k0+23};
//                 lane>=16 -> row=L-16, K={k0+8..k0+15, k0+24..k0+31}.
// ---------------------------------------------------------------------------
__global__ void prep_x_bf16(const float* __restrict__ x, __bf16* __restrict__ out,
                            int nFrag) {
    int gid = blockIdx.x * blockDim.x + threadIdx.x;
    int f = gid >> 5;
    if (f >= nFrag) return;
    int l     = gid & 31;
    int mtile = f / 24;
    int kc    = f - mtile * 24;
    int row   = mtile * 16 + (l & 15);
    int kb    = kc * 32 + (l >> 4) * 8;   // first 8-elem run
    const f32x4* s = reinterpret_cast<const f32x4*>(x + (size_t)row * I_ + kb);
    float buf[16];
#pragma unroll
    for (int q = 0; q < 2; ++q) {       // k run 1: kb .. kb+7
        f32x4 v = s[q];
        buf[q*4+0]=v.x; buf[q*4+1]=v.y; buf[q*4+2]=v.z; buf[q*4+3]=v.w;
    }
    s = reinterpret_cast<const f32x4*>(x + (size_t)row * I_ + kb + 16);
#pragma unroll
    for (int q = 0; q < 2; ++q) {       // k run 2: kb+16 .. kb+23
        f32x4 v = s[q];
        buf[8+q*4+0]=v.x; buf[8+q*4+1]=v.y; buf[8+q*4+2]=v.z; buf[8+q*4+3]=v.w;
    }
    FragU fr;
#pragma unroll
    for (int j = 0; j < 16; ++j) fr.h[j] = (__bf16)buf[j];
    f32x4* dst = reinterpret_cast<f32x4*>(out + (size_t)f * 512 + l * 16);
    dst[0] = fr.f4[0];
    dst[1] = fr.f4[1];
}

// ---------------------------------------------------------------------------
// Kernel 1: xg[B*S,3H] = x[B*S,I] @ W_ih^T + b_ih  (bf16 WMMA, f32 accum)
// 256 threads (8 waves), M-tile 128 (wave = 16 rows), N in chunks of 64
// (4 tiles/wave), K in steps of 32. The pre-packed bf16 A tile is DMA'd
// into a double-buffered LDS stage with GLOBAL_LOAD_ASYNC_TO_LDS_B128 so
// the async fill of chunk kt+1 overlaps the WMMAs of chunk kt.
// ---------------------------------------------------------------------------
__global__ void __launch_bounds__(256)
gemm_xg(const __bf16* __restrict__ xb, const float* __restrict__ b_ih,
        const __bf16* __restrict__ wih, float* __restrict__ xg) {
    __shared__ f32x4 aBuf[2][512];     // 2 x 8KB A-fragment stages

    const int tid = threadIdx.x;
    const int wm  = tid >> 5;          // wave id = M sub-tile
    const int l   = tid & 31;
    const size_t mt0 = (size_t)blockIdx.x * 8;   // first 16-row M-tile
    const f32x4* w4  = reinterpret_cast<const f32x4*>(wih);
    const char*  xbc = reinterpret_cast<const char*>(xb);

    // per-thread async source/dest: frag (mt0+wm)*24 + kt, 32B slice at l*32
    const size_t srcBase = ((mt0 + wm) * 24) * 1024 + (size_t)l * 32;

    auto issue = [&](int kt, int buf) {
        const char* s = xbc + srcBase + (size_t)kt * 1024;
        char*       d = reinterpret_cast<char*>(&aBuf[buf][0]) + wm * 1024 + l * 32;
        async_b128(s, d);
        async_b128(s + 16, d + 16);
    };

    for (int nc = 0; nc < 12; ++nc) {
        v8f acc[4];
        v8f zero = {0.f,0.f,0.f,0.f,0.f,0.f,0.f,0.f};
#pragma unroll
        for (int ti = 0; ti < 4; ++ti) acc[ti] = zero;

        issue(0, 0);                    // prologue fill of buffer 0
        for (int kt = 0; kt < 24; ++kt) {
            wait_async0();              // my copies for chunk kt landed
            __syncthreads();            // everyone's copies landed, prior reads done
            if (kt + 1 < 24) issue(kt + 1, (kt + 1) & 1);
            FragU A;
            A.f4[0] = aBuf[kt & 1][wm*64 + l*2];
            A.f4[1] = aBuf[kt & 1][wm*64 + l*2 + 1];
#pragma unroll
            for (int ti = 0; ti < 4; ++ti) {
                int nt = nc*4 + ti;
                const f32x4* bp = w4 + (size_t)(nt*24 + kt)*64 + l*2;
                FragU Bf;
                Bf.f4[0] = bp[0];
                Bf.f4[1] = bp[1];
                acc[ti] = __builtin_amdgcn_wmma_f32_16x16x32_bf16(
                    false, A.v, false, Bf.v, (short)0, acc[ti], false, false);
            }
        }
        // ---- epilogue: + bias, store f32 ----
        int rowOff = (l & 16) ? 8 : 0;
#pragma unroll
        for (int ti = 0; ti < 4; ++ti) {
            int col = nc*64 + ti*16 + (l & 15);
            float bias = b_ih[col];
#pragma unroll
            for (int v = 0; v < 8; ++v) {
                size_t row = mt0*16 + wm*16 + v + rowOff;
                xg[row * G3 + col] = acc[ti][v] + bias;
            }
        }
    }
}

// ---------------------------------------------------------------------------
// Kernel 2: the sequential scan. 8 workgroups x 256 threads; WG owns 16
// batch rows for all 1024 steps (batch-independent recurrence). Per step:
//   A) decay h by exp(-|alpha|*dt), re-encode h as bf16 A-fragments in LDS,
//      prefetch next step's xg tile so phase C hits warm cache
//   B) hg = h @ W_hh^T via 384 WMMAs (wave w owns cols [32w,32w+32) of ALL
//      three gates -> r/z/n align lane-for-lane in the accumulators)
//   C) gates + h update fully in-register (v_exp/v_rcp), h kept in LDS f32
//   D) out[:,t,:] = h @ W_fc^T + b_fc  (O=6, VALU dot products)
// ---------------------------------------------------------------------------
__global__ void __launch_bounds__(256)
scan_gru(const float* __restrict__ xg, const __bf16* __restrict__ whh,
         const float* __restrict__ rel_pos, const float* __restrict__ alpha,
         const float* __restrict__ b_hh, const float* __restrict__ W_fc,
         const float* __restrict__ b_fc, float* __restrict__ out) {
    __shared__ float hLds[16 * 256];   // decayed/updated hidden state (16KB)
    __shared__ f32x4 aLds4[512];       // h as bf16 A fragments, 8 K-chunks (8KB)
    __shared__ float wfc[O_ * H_];     // 6KB
    __shared__ float bfc[8];

    const int tid = threadIdx.x;
    const int w   = tid >> 5;
    const int l   = tid & 31;
    const int b0  = blockIdx.x * 16;

    for (int i = tid; i < 16*256; i += 256) hLds[i] = 0.0f;
    for (int i = tid; i < O_*H_;  i += 256) wfc[i] = W_fc[i];
    if (tid < O_) bfc[tid] = b_fc[tid];

    const float a = fabsf(alpha[0]);

    // per-lane gate bias constants (column assignment is step-invariant)
    float bhh_c[6];
#pragma unroll
    for (int sub = 0; sub < 2; ++sub) {
        int c = 32*w + sub*16 + (l & 15);
        bhh_c[sub]     = b_hh[c];
        bhh_c[2 + sub] = b_hh[256 + c];
        bhh_c[4 + sub] = b_hh[512 + c];
    }
    const int bA     = tid >> 4;          // phase-A batch row (0..15)
    const int c0     = (tid & 15) * 16;   // phase-A column base
    const int kcA    = c0 >> 5;
    const int halfkA = (c0 >> 4) & 1;
    const f32x4* w4  = reinterpret_cast<const f32x4*>(whh);

    __syncthreads();

    for (int t = 0; t < S_; ++t) {
        // ---- Phase A: time decay + bf16 fragment encode + xg prefetch ----
        {
            size_t rpi = (size_t)(b0 + bA) * S_ + t;
            float rp  = rel_pos[rpi];
            float rpm = (t == 0) ? 0.0f : rel_pos[rpi - 1];
            float dec = __expf(-a * (rp - rpm));
            FragU fr;
#pragma unroll
            for (int j = 0; j < 16; ++j) {
                float hv = hLds[bA*256 + c0 + j] * dec;
                hLds[bA*256 + c0 + j] = hv;
                fr.h[j] = (__bf16)hv;
            }
            aLds4[kcA*64 + bA*2      + halfkA] = fr.f4[0];
            aLds4[kcA*64 + (bA+16)*2 + halfkA] = fr.f4[1];
            if (t + 1 < S_) {   // warm next step's 48KB xg tile behind phase B
                size_t xn = ((size_t)(b0 + bA) * S_ + (t + 1)) * (size_t)G3 + c0;
                __builtin_prefetch(&xg[xn],       0, 0);
                __builtin_prefetch(&xg[xn + 256], 0, 0);
                __builtin_prefetch(&xg[xn + 512], 0, 0);
            }
        }
        __syncthreads();
        // ---- Phase B: WMMA recurrence GEMM ; Phase C: gates in-register ----
        {
            v8f acc[6];
            v8f zero = {0.f,0.f,0.f,0.f,0.f,0.f,0.f,0.f};
#pragma unroll
            for (int s = 0; s < 6; ++s) acc[s] = zero;
#pragma unroll
            for (int kc = 0; kc < 8; ++kc) {
                FragU A;
                A.f4[0] = aLds4[kc*64 + l*2];
                A.f4[1] = aLds4[kc*64 + l*2 + 1];
#pragma unroll
                for (int s = 0; s < 6; ++s) {
                    int nt = (s < 2) ? (2*w + s)
                           : (s < 4) ? (16 + 2*w + (s - 2))
                                     : (32 + 2*w + (s - 4));
                    const f32x4* bp = w4 + (size_t)(nt*8 + kc)*64 + l*2;
                    FragU Bf;
                    Bf.f4[0] = bp[0];
                    Bf.f4[1] = bp[1];
                    acc[s] = __builtin_amdgcn_wmma_f32_16x16x32_bf16(
                        false, A.v, false, Bf.v, (short)0, acc[s], false, false);
                }
            }
            int rowOff = (l & 16) ? 8 : 0;
#pragma unroll
            for (int sub = 0; sub < 2; ++sub) {
                int c = 32*w + sub*16 + (l & 15);
#pragma unroll
                for (int v = 0; v < 8; ++v) {
                    int row = v + rowOff;
                    size_t xi = ((size_t)(b0 + row) * S_ + t) * (size_t)G3 + c;
                    float xr = xg[xi];
                    float xz = xg[xi + 256];
                    float xn = xg[xi + 512];
                    float hr = acc[sub][v]     + bhh_c[sub];
                    float hz = acc[2 + sub][v] + bhh_c[2 + sub];
                    float hn = acc[4 + sub][v] + bhh_c[4 + sub];
                    float r = sigmoid_f(xr + hr);
                    float z = sigmoid_f(xz + hz);
                    float n = tanh_f(xn + r * hn);
                    float hold = hLds[row*256 + c];
                    hLds[row*256 + c] = (1.0f - z) * n + z * hold;
                }
            }
        }
        __syncthreads();
        // ---- Phase D: output projection (O = 6) ----
        if (tid < 96) {
            int b = tid / 6, o = tid - b * 6;
            float accO = bfc[o];
            for (int c = 0; c < H_; ++c)
                accO += wfc[o*H_ + c] * hLds[b*256 + c];
            out[((size_t)(b0 + b) * S_ + t) * O_ + o] = accO;
        }
        __syncthreads();
    }
}

// ---------------------------------------------------------------------------
extern "C" void kernel_launch(void* const* d_in, const int* in_sizes, int n_in,
                              void* d_out, int out_size, void* d_ws, size_t ws_size,
                              hipStream_t stream) {
    const float* x       = (const float*)d_in[0];
    const float* rel_pos = (const float*)d_in[1];
    const float* alpha   = (const float*)d_in[2];
    const float* W_ih    = (const float*)d_in[3];
    const float* b_ih    = (const float*)d_in[4];
    const float* W_hh    = (const float*)d_in[5];
    const float* b_hh    = (const float*)d_in[6];
    const float* W_fc    = (const float*)d_in[7];
    const float* b_fc    = (const float*)d_in[8];
    float* outp          = (float*)d_out;

    // workspace layout
    const size_t XG_BYTES  = (size_t)B_ * S_ * G3 * sizeof(float);    // 402.7 MB
    const size_t XB_BYTES  = (size_t)B_ * S_ * I_ * sizeof(__bf16);   // 201.3 MB
    const size_t WIH_BYTES = (size_t)G3 * I_ * sizeof(__bf16);        // 1.18 MB
    float*  xgws = (float*)d_ws;
    __bf16* xb   = (__bf16*)((char*)d_ws + XG_BYTES);
    __bf16* wih  = (__bf16*)((char*)d_ws + XG_BYTES + XB_BYTES);
    __bf16* whh  = (__bf16*)((char*)d_ws + XG_BYTES + XB_BYTES + WIH_BYTES);

    // pack activations + weights as bf16 WMMA fragments
    const int nFragX = (B_ * S_ / 16) * 24;                    // 196608
    prep_x_bf16<<<nFragX * 32 / 256, 256, 0, stream>>>(x, xb, nFragX);
    prep_w_bf16<<<144, 256, 0, stream>>>(W_ih, wih, I_, 24, 48 * 24);
    prep_w_bf16<<<48,  256, 0, stream>>>(W_hh, whh, H_,  8, 48 * 8);

    // batched input projection (memory-bound GEMM, async-to-LDS pipelined)
    gemm_xg<<<B_ * S_ / 128, 256, 0, stream>>>(xb, b_ih, wih, xgws);

    // sequential time-decayed GRU scan + output head
    scan_gru<<<B_ / 16, 256, 0, stream>>>(xgws, whh, rel_pos, alpha,
                                          b_hh, W_fc, b_fc, outp);
}
